// FeatureExtraction_15290083573839
// MI455X (gfx1250) — compile-verified, matrix operand order checked by reading, not run on hardware
//
#include <hip/hip_runtime.h>
#include <hip/hip_bf16.h>
#include <stdint.h>

typedef __attribute__((ext_vector_type(16))) _Float16 v16h;
typedef __attribute__((ext_vector_type(8)))  float    v8f;

static constexpr int Bn = 8;
static constexpr int Nn = 2048;
static constexpr int Kk = 32;

// ---------------------------------------------------------------------------
// kNN: one workgroup (256 threads) per query point. Exact fp32 distances in
// LDS, then 33 rounds of argmin with (float_bits<<32 | index) packed keys
// (monotonic for d2 >= 0; ties -> smaller index, matching top_k). Round 0
// removes the self point (distance exactly 0), rounds 1..32 emit neighbors.
// ---------------------------------------------------------------------------
template <int C>
__global__ void knn_kernel(const float* __restrict__ x, int* __restrict__ idx_out) {
  const int bn  = blockIdx.x;
  const int b   = bn / Nn, n = bn % Nn;
  const int tid = threadIdx.x;              // 256 threads

  __shared__ float xq[C];
  __shared__ float dist[Nn];
  __shared__ unsigned long long red[256];

  const float* xb = x + (size_t)b * Nn * C;
  for (int c = tid; c < C; c += 256) xq[c] = xb[(size_t)n * C + c];
  __syncthreads();

  for (int m = tid; m < Nn; m += 256) {
    const float* xm = xb + (size_t)m * C;
    float d = 0.f;
#pragma unroll 4
    for (int c = 0; c < C; ++c) { float df = xq[c] - xm[c]; d += df * df; }
    dist[m] = d;
  }
  __syncthreads();

  for (int r = 0; r <= Kk; ++r) {
    unsigned long long best = ~0ull;
    for (int m = tid; m < Nn; m += 256) {
      unsigned long long key =
          ((unsigned long long)__float_as_uint(dist[m]) << 32) | (unsigned)m;
      best = key < best ? key : best;
    }
    red[tid] = best;
    __syncthreads();
#pragma unroll
    for (int s = 128; s > 0; s >>= 1) {
      if (tid < s) {
        unsigned long long o = red[tid + s];
        if (o < red[tid]) red[tid] = o;
      }
      __syncthreads();
    }
    if (tid == 0) {
      int w = (int)(red[0] & 0xffffffffu);
      dist[w] = 3.4e38f;
      if (r > 0) idx_out[((size_t)b * Nn + n) * Kk + (r - 1)] = w;
    }
    __syncthreads();
  }
}

// ---------------------------------------------------------------------------
// Pre-swizzle W (fp32 [realK x Cout]) into f16 B-fragment order so each lane
// loads its 16 halves (32B) contiguously per (tile t, k-chunk q).
// Layout: Wswz[(((t*NQ + q)*32 + lane)*16 + h)]; lanes 0-15 hold column
// n=t*16+lane with halves K = q*32 + {0..7, 16..23}; lanes 16-31 hold
// K = q*32 + {8..15, 24..31} (mirrors the documented 16-bit A layout).
// Rows kk >= realK are zero-padded (stage 1: 6 -> 32).
// ---------------------------------------------------------------------------
__global__ void wprep_kernel(const float* __restrict__ W, _Float16* __restrict__ wswz,
                             int realK, int Cout, int Kpad) {
  const int total = Cout * Kpad;
  const int nq = Kpad / 32;
  for (int e = blockIdx.x * blockDim.x + threadIdx.x; e < total;
       e += gridDim.x * blockDim.x) {
    int h    = e & 15;
    int lane = (e >> 4) & 31;
    int qt   = e >> 9;
    int q    = qt % nq;
    int t    = qt / nq;
    int nCol = t * 16 + (lane & 15);
    int kk   = q * 32 + ((lane < 16) ? ((h < 8) ? h : 16 + (h - 8))
                                     : ((h < 8) ? 8 + h : 24 + (h - 8)));
    float v = (kk < realK) ? W[(size_t)kk * Cout + nCol] : 0.f;
    wswz[e] = (_Float16)v;
  }
}

__global__ void concat_kernel(const float* __restrict__ x1, const float* __restrict__ x2,
                              float* __restrict__ xc) {
  const int total = Bn * Nn * 192;
  for (int e = blockIdx.x * blockDim.x + threadIdx.x; e < total;
       e += gridDim.x * blockDim.x) {
    int p = e / 192, c = e % 192;
    xc[e] = (c < 64) ? x1[(size_t)p * 64 + c] : x2[(size_t)p * 128 + (c - 64)];
  }
}

// ---------------------------------------------------------------------------
// Fused EdgeConv: one workgroup (4 waves) per point n. Build the 32 message
// rows [xi, xj-xi] in LDS (f16), then each wave produces 16-col output tiles:
// two v_wmma_f32_16x16x32_f16 row-tiles (k=0..15 / 16..31) per k-chunk,
// max over k via per-lane max + shfl_xor(16), fused bias (+relu).
// relu/bias commute with max_k, so max is taken on raw accumulators.
// ---------------------------------------------------------------------------
template <int CIN, int KPAD, int COUT, bool RELU>
__global__ void edgeconv_kernel(const float* __restrict__ feat, const int* __restrict__ idx,
                                const _Float16* __restrict__ wswz,
                                const float* __restrict__ bias, float* __restrict__ out) {
  constexpr int NQ = KPAD / 32;
  constexpr int NT = COUT / 16;
  const int bn   = blockIdx.x;
  const int b    = bn / Nn, n = bn % Nn;
  const int tid  = threadIdx.x;            // 128 threads = 4 waves
  const int lane = tid & 31, wave = tid >> 5;

  __shared__ _Float16 msg[32 * KPAD];
  __shared__ int nbr[32];

  const float* fb = feat + (size_t)b * Nn * CIN;
  const float* xi = fb + (size_t)n * CIN;
  if (tid < 32) nbr[tid] = idx[((size_t)b * Nn + n) * Kk + tid];
  __syncthreads();

  for (int e = tid; e < 32 * KPAD; e += 128) {
    int k = e / KPAD;
    int c = e % KPAD;
    float v;
    if (c < CIN)            v = xi[c];
    else if (c < 2 * CIN) { int cc = c - CIN; v = fb[(size_t)nbr[k] * CIN + cc] - xi[cc]; }
    else                    v = 0.f;
    msg[e] = (_Float16)v;
  }
  __syncthreads();

  const int m0   = lane & 15;
  const int half = (lane < 16) ? 0 : 8;

  for (int t = wave; t < NT; t += 4) {
    v8f acc0 = {}, acc1 = {};
#pragma unroll
    for (int q = 0; q < NQ; ++q) {
      v16h bf = *(const v16h*)(wswz + ((size_t)(t * NQ + q) * 32 + lane) * 16);
      const _Float16* p0 = &msg[m0 * KPAD + q * 32 + half];
      const _Float16* p1 = &msg[(m0 + 16) * KPAD + q * 32 + half];
      union { v16h v; _Float16 h[16]; } a0, a1;
#pragma unroll
      for (int i = 0; i < 8; ++i) {
        a0.h[i] = p0[i];  a0.h[8 + i] = p0[16 + i];
        a1.h[i] = p1[i];  a1.h[8 + i] = p1[16 + i];
      }
      acc0 = __builtin_amdgcn_wmma_f32_16x16x32_f16(false, a0.v, false, bf,
                                                    (short)0, acc0, false, false);
      acc1 = __builtin_amdgcn_wmma_f32_16x16x32_f16(false, a1.v, false, bf,
                                                    (short)0, acc1, false, false);
    }
    float m = acc0[0];
#pragma unroll
    for (int i = 1; i < 8; ++i) m = fmaxf(m, acc0[i]);
#pragma unroll
    for (int i = 0; i < 8; ++i) m = fmaxf(m, acc1[i]);
    m = fmaxf(m, __shfl_xor(m, 16, 32));     // combine the two M-halves (lane ^ 16)
    if (lane < 16) {
      int d  = t * 16 + lane;
      float r = m + bias[d];
      if (RELU) r = fmaxf(r, 0.f);
      out[((size_t)b * Nn + n) * COUT + d] = r;
    }
  }
}

// ---------------------------------------------------------------------------
extern "C" void kernel_launch(void* const* d_in, const int* in_sizes, int n_in,
                              void* d_out, int out_size, void* d_ws, size_t ws_size,
                              hipStream_t stream) {
  const float* x  = (const float*)d_in[0];
  const float* W1 = (const float*)d_in[1];
  const float* b1 = (const float*)d_in[2];
  const float* W2 = (const float*)d_in[3];
  const float* b2 = (const float*)d_in[4];
  const float* W3 = (const float*)d_in[5];
  const float* b3 = (const float*)d_in[6];

  char* ws = (char*)d_ws;
  size_t off = 0;
  auto alloc = [&](size_t bytes) { void* p = ws + off; off = (off + bytes + 255) & ~(size_t)255; return p; };

  int*       idx   = (int*)      alloc((size_t)Bn * Nn * Kk * sizeof(int));     // 2 MB
  float*     x1    = (float*)    alloc((size_t)Bn * Nn * 64  * sizeof(float));  // 4 MB
  float*     x2    = (float*)    alloc((size_t)Bn * Nn * 128 * sizeof(float));  // 8 MB
  float*     xc    = (float*)    alloc((size_t)Bn * Nn * 192 * sizeof(float));  // 12 MB
  _Float16*  w1swz = (_Float16*) alloc((size_t)64  * 32  * sizeof(_Float16));
  _Float16*  w2swz = (_Float16*) alloc((size_t)128 * 128 * sizeof(_Float16));
  _Float16*  w3swz = (_Float16*) alloc((size_t)512 * 384 * sizeof(_Float16));
  float*     outF  = (float*)d_out;

  const int BN = Bn * Nn;

  // Weight pre-swizzle into WMMA B-fragment layout (f16)
  wprep_kernel<<<8,   256, 0, stream>>>(W1, w1swz,   6,  64,  32);
  wprep_kernel<<<64,  256, 0, stream>>>(W2, w2swz, 128, 128, 128);
  wprep_kernel<<<256, 256, 0, stream>>>(W3, w3swz, 384, 512, 384);

  // Stage 1
  knn_kernel<3><<<BN, 256, 0, stream>>>(x, idx);
  edgeconv_kernel<3, 32, 64, true><<<BN, 128, 0, stream>>>(x, idx, w1swz, b1, x1);

  // Stage 2
  knn_kernel<64><<<BN, 256, 0, stream>>>(x1, idx);
  edgeconv_kernel<64, 128, 128, true><<<BN, 128, 0, stream>>>(x1, idx, w2swz, b2, x2);

  // Stage 3
  knn_kernel<128><<<BN, 256, 0, stream>>>(x2, idx);
  concat_kernel<<<512, 256, 0, stream>>>(x1, x2, xc);
  edgeconv_kernel<192, 384, 512, false><<<BN, 128, 0, stream>>>(xc, idx, w3swz, b3, outF);
}
